// SentiQDNN_71038759076456
// MI455X (gfx1250) — compile-verified, matrix operand order checked by reading, not run on hardware
//
#include <hip/hip_runtime.h>
#include <hip/hip_bf16.h>
#include <math.h>
#include <stdint.h>

// Problem constants (match reference)
#define NB   64
#define NS   256
#define ND   512
#define NU   256
#define NTOK (NB * NS)   // 16384
#define LDRI 1024        // RI row stride: [re(512) | im(512)]

typedef __attribute__((ext_vector_type(16))) __bf16       v16bf;
typedef __attribute__((ext_vector_type(8)))  __bf16       v8bf;
typedef __attribute__((ext_vector_type(8)))  float        v8f;
typedef __attribute__((ext_vector_type(8)))  unsigned int v8u;
typedef __attribute__((ext_vector_type(4)))  unsigned int v4u;
typedef __attribute__((ext_vector_type(8)))  int          v8i_t;
typedef __attribute__((ext_vector_type(4)))  int          v4i_t;

#if __has_builtin(__builtin_amdgcn_tensor_load_to_lds)
#define USE_TDM 1
#else
#define USE_TDM 0
#endif

// ---------------------------------------------------------------------------
// Fragment load following the documented CDNA5 16-bit A-matrix layout:
//   lanes 0-15: row M = lane,  VGPR0..3 = K 0..7,  VGPR4..7 = K 16..23
//   lanes16-31: row M = lane-16, VGPR0..3 = K 8..15, VGPR4..7 = K 24..31
// => per lane two 16B loads at (row, k + hi*8) and (row, k + hi*8 + 16).
// Works for global (A: row=token) and LDS (B: row=unit) operands.
// ---------------------------------------------------------------------------
__device__ __forceinline__ v16bf ldfrag(const __bf16* p) {
    v8bf lo = *(const v8bf*)(p);
    v8bf hi = *(const v8bf*)(p + 16);
    v16bf f;
#pragma unroll
    for (int i = 0; i < 8; ++i) { f[i] = lo[i]; f[i + 8] = hi[i]; }
    return f;
}

__device__ __forceinline__ v16bf negbf(v16bf x) {  // flip bf16 sign bits
    v8u u = __builtin_bit_cast(v8u, x);
    u = u ^ 0x80008000u;
    return __builtin_bit_cast(v16bf, u);
}

#if USE_TDM
// ---------------------------------------------------------------------------
// TDM: 2D tile load Global->LDS per ISA D# layout (CDNA5 ISA).
//   group0: [1:0]=count=1 | [63:32]=lds_addr | [120:64]=global_addr | type=2
//   group1: data_size | tensor_dim0/1 | tile_dim0/1 | tensor_dim0_stride
// data_size=1 => 2-byte elements (bf16).
// ---------------------------------------------------------------------------
__device__ __forceinline__ void tdm_load_2d(unsigned int lds_addr,
                                            unsigned long long gaddr,
                                            unsigned int tensor_d0,
                                            unsigned int tensor_d1,
                                            unsigned int tile_d0,
                                            unsigned int tile_d1,
                                            unsigned long long stride0)
{
    v4u g0;
    g0[0] = 1u;                                          // count=1, user mode
    g0[1] = lds_addr;                                    // LDS byte address
    g0[2] = (unsigned int)gaddr;                         // global_addr[31:0]
    g0[3] = ((unsigned int)(gaddr >> 32) & 0x01FFFFFFu)  // global_addr[56:32]
            | 0x80000000u;                               // type=2 ("image")
    v8i_t g1;
    g1[0] = (int)(1u << 16);                             // data_size=1 (2B)
    g1[1] = (int)((tensor_d0 & 0xFFFFu) << 16);          // tensor_dim0[15:0]
    g1[2] = (int)((tensor_d0 >> 16) | ((tensor_d1 & 0xFFFFu) << 16));
    g1[3] = (int)((tensor_d1 >> 16) | ((tile_d0 & 0xFFFFu) << 16));
    g1[4] = (int)(tile_d1 & 0xFFFFu);                    // tile_dim1, tile_dim2=0
    g1[5] = (int)(unsigned int)(stride0 & 0xFFFFFFFFull);// dim0_stride[31:0]
    g1[6] = (int)((unsigned int)(stride0 >> 32) & 0xFFFFu);
    g1[7] = 0;                                           // dim1_stride unused (2D)
    v4i_t z4 = {0, 0, 0, 0};
#if defined(__clang_major__) && (__clang_major__ >= 23)
    v8i_t z8 = {0, 0, 0, 0, 0, 0, 0, 0};
    __builtin_amdgcn_tensor_load_to_lds(g0, g1, z4, z4, z8, 0);
#else
    __builtin_amdgcn_tensor_load_to_lds(g0, g1, z4, z4, 0);
#endif
}
#endif  // USE_TDM

// ---------------------------------------------------------------------------
// 1) Gather amp/phase per token, L2-normalize amp, re/im = amp_n * e^{i phase}
// ---------------------------------------------------------------------------
__global__ __launch_bounds__(256) void embed_kernel(
    const int* __restrict__ seq, const int* __restrict__ lex,
    const float* __restrict__ ampT, const float* __restrict__ phT,
    __bf16* __restrict__ RI, float* __restrict__ norms,
    long long* __restrict__ tag_out)
{
    const int t   = blockIdx.x;            // token id
    const int tid = threadIdx.x;           // 256 threads, 2 d-elems each
    const int tok = seq[t];

    const float2 a  = ((const float2*)(ampT + (size_t)tok * ND))[tid];
    const float2 ph = ((const float2*)(phT  + (size_t)tok * ND))[tid];

    __shared__ float red[256];
    red[tid] = a.x * a.x + a.y * a.y;
    __syncthreads();
#pragma unroll
    for (int s = 128; s > 0; s >>= 1) {
        if (tid < s) red[tid] += red[tid + s];
        __syncthreads();
    }
    const float nrm = sqrtf(red[0]);
    if (tid == 0) {
        norms[t]   = nrm;
        tag_out[t] = (long long)lex[tok];
    }
    const float inv = 1.0f / fmaxf(nrm, 1e-12f);

    __bf16* row = RI + (size_t)t * LDRI;
    const int d = tid * 2;
    row[d]          = (__bf16)(a.x * inv * cosf(ph.x));
    row[d + 1]      = (__bf16)(a.y * inv * cosf(ph.y));
    row[ND + d]     = (__bf16)(a.x * inv * sinf(ph.x));
    row[ND + d + 1] = (__bf16)(a.y * inv * sinf(ph.y));
}

// ---------------------------------------------------------------------------
// 2) Softmax over sequence dim of norms -> w[b,s]; also zero measured[b,u].
// ---------------------------------------------------------------------------
__global__ __launch_bounds__(256) void softmax_kernel(
    const float* __restrict__ norms, float* __restrict__ w,
    float* __restrict__ measured)
{
    const int b = blockIdx.x, s = threadIdx.x;
    const float x = norms[b * NS + s];

    __shared__ float red[256];
    red[s] = x;
    __syncthreads();
#pragma unroll
    for (int k = 128; k > 0; k >>= 1) {
        if (s < k) red[s] = fmaxf(red[s], red[s + k]);
        __syncthreads();
    }
    const float mx = red[0];
    __syncthreads();
    const float e = __expf(x - mx);
    red[s] = e;
    __syncthreads();
#pragma unroll
    for (int k = 128; k > 0; k >>= 1) {
        if (s < k) red[s] += red[s + k];
        __syncthreads();
    }
    w[b * NS + s] = e / red[0];
    measured[b * NU + s] = 0.0f;           // NU == NS == 256: exact cover
}

// ---------------------------------------------------------------------------
// 3) Normalize measurement kernel; store unit-major bf16 Kr[U,D], Ki[U,D].
// ---------------------------------------------------------------------------
__global__ __launch_bounds__(256) void prepk_kernel(
    const float* __restrict__ mk, __bf16* __restrict__ Kr,
    __bf16* __restrict__ Ki)
{
    const int u   = blockIdx.x;
    const int tid = threadIdx.x;           // 4 floats (2 complex elems) each
    const float4 v = ((const float4*)(mk + (size_t)u * ND * 2))[tid];

    __shared__ float red[256];
    red[tid] = v.x * v.x + v.y * v.y + v.z * v.z + v.w * v.w;
    __syncthreads();
#pragma unroll
    for (int s = 128; s > 0; s >>= 1) {
        if (tid < s) red[tid] += red[tid + s];
        __syncthreads();
    }
    const float inv = 1.0f / fmaxf(sqrtf(red[0]), 1e-12f);

    const int d = tid * 2;
    Kr[(size_t)u * ND + d]     = (__bf16)(v.x * inv);
    Ki[(size_t)u * ND + d]     = (__bf16)(v.y * inv);
    Kr[(size_t)u * ND + d + 1] = (__bf16)(v.z * inv);
    Ki[(size_t)u * ND + d + 1] = (__bf16)(v.w * inv);
}

// ---------------------------------------------------------------------------
// 4) Fused WMMA GEMM + measurement.
//    P = re*Kr + im*Ki ; Q = im*Kr - re*Ki ; measured[b,u] += Σ_rows w·(P²+Q²)
//    Block = 128 threads = 4 waves; wave owns one 16-token x 16-unit tile.
//    B operand (Kr/Ki tiles for the block's 64 units) staged into LDS by the
//    Tensor Data Mover, double-buffered, paced with s_wait_tensorcnt.
//    A operand streams from global (b128 fragments + prefetch).
// ---------------------------------------------------------------------------
__global__ __launch_bounds__(128) void qmeasure_wmma_kernel(
    const __bf16* __restrict__ RI, const __bf16* __restrict__ Kr,
    const __bf16* __restrict__ Ki, const float* __restrict__ w,
    float* __restrict__ measured)
{
    const int wave = threadIdx.x >> 5;
    const int lane = threadIdx.x & 31;
    const int row0 = blockIdx.x * 16;                 // token tile base
    const int col0 = blockIdx.y * 64 + wave * 16;     // unit  tile base

    const int r  = lane & 15;
    const int hi = lane >> 4;

    const __bf16* pre = RI + (size_t)(row0 + r) * LDRI + hi * 8;   // re half
    const __bf16* pim = pre + ND;                                   // im half

    v8f P = {};
    v8f Q = {};

#if USE_TDM
    // [buf][kr=0/ki=1][unit 0..63][k 0..31], 16 KB total, double buffered
    __shared__ __align__(1024) __bf16 bst[2][2][64][32];

    const unsigned long long gKr =
        (unsigned long long)(uintptr_t)(Kr + (size_t)blockIdx.y * 64 * ND);
    const unsigned long long gKi =
        (unsigned long long)(uintptr_t)(Ki + (size_t)blockIdx.y * 64 * ND);

    if (wave == 0) {   // prologue: stage k-chunk 0 into buffer 0
        tdm_load_2d((unsigned int)(uintptr_t)&bst[0][0][0][0], gKr,
                    ND, 64, 32, 64, ND);
        tdm_load_2d((unsigned int)(uintptr_t)&bst[0][1][0][0], gKi,
                    ND, 64, 32, 64, ND);
    }

    for (int kc = 0; kc < ND / 32; ++kc) {
        const int k   = kc * 32;
        const int buf = kc & 1;

        if (wave == 0) {
            if (kc + 1 < ND / 32) {   // stage next chunk into other buffer
                tdm_load_2d((unsigned int)(uintptr_t)&bst[buf ^ 1][0][0][0],
                            gKr + (size_t)(k + 32) * 2, ND, 64, 32, 64, ND);
                tdm_load_2d((unsigned int)(uintptr_t)&bst[buf ^ 1][1][0][0],
                            gKi + (size_t)(k + 32) * 2, ND, 64, 32, 64, ND);
                __builtin_amdgcn_s_wait_tensorcnt(2);  // chunk kc complete
            } else {
                __builtin_amdgcn_s_wait_tensorcnt(0);
            }
        }
        __syncthreads();              // staged data visible to all 4 waves

        const v16bf a_re = ldfrag(pre + k);
        const v16bf a_im = ldfrag(pim + k);
        if (k + 32 < ND) {
            __builtin_prefetch(pre + k + 32, 0, 3);
            __builtin_prefetch(pim + k + 32, 0, 3);
        }
        const v16bf b_kr = ldfrag(&bst[buf][0][wave * 16 + r][hi * 8]);
        const v16bf b_ki = ldfrag(&bst[buf][1][wave * 16 + r][hi * 8]);

        P = __builtin_amdgcn_wmma_f32_16x16x32_bf16(false, a_re, false, b_kr,
                                                    (short)0, P, false, false);
        Q = __builtin_amdgcn_wmma_f32_16x16x32_bf16(false, a_im, false, b_kr,
                                                    (short)0, Q, false, false);
        P = __builtin_amdgcn_wmma_f32_16x16x32_bf16(false, a_im, false, b_ki,
                                                    (short)0, P, false, false);
        const v16bf a_rn = negbf(a_re);  // bf16 WMMA: NEG[1:0] must be 0
        Q = __builtin_amdgcn_wmma_f32_16x16x32_bf16(false, a_rn, false, b_ki,
                                                    (short)0, Q, false, false);

        __syncthreads();              // buffer reuse fence
    }
#else
    const __bf16* pkr = Kr + (size_t)(col0 + r) * ND + hi * 8;
    const __bf16* pki = Ki + (size_t)(col0 + r) * ND + hi * 8;
#pragma unroll 4
    for (int k = 0; k < ND; k += 32) {
        const v16bf a_re = ldfrag(pre + k);
        const v16bf a_im = ldfrag(pim + k);
        const v16bf b_kr = ldfrag(pkr + k);
        const v16bf b_ki = ldfrag(pki + k);
        P = __builtin_amdgcn_wmma_f32_16x16x32_bf16(false, a_re, false, b_kr,
                                                    (short)0, P, false, false);
        Q = __builtin_amdgcn_wmma_f32_16x16x32_bf16(false, a_im, false, b_kr,
                                                    (short)0, Q, false, false);
        P = __builtin_amdgcn_wmma_f32_16x16x32_bf16(false, a_im, false, b_ki,
                                                    (short)0, P, false, false);
        const v16bf a_rn = negbf(a_re);
        Q = __builtin_amdgcn_wmma_f32_16x16x32_bf16(false, a_rn, false, b_ki,
                                                    (short)0, Q, false, false);
    }
#endif

    // C/D layout: lanes 0-15 -> N=lane, M=vgpr; lanes 16-31 -> N=lane-16, M=vgpr+8
    float acc = 0.0f;
#pragma unroll
    for (int m = 0; m < 8; ++m) {
        const float wv = w[row0 + m + hi * 8];
        acc = fmaf(wv, P[m] * P[m] + Q[m] * Q[m], acc);
    }
    acc += __shfl_xor(acc, 16, 32);             // combine M halves per column
    if (lane < 16) {
        const int b = row0 / NS;                // 16-row tiles never cross b
        atomicAdd(&measured[b * NU + col0 + lane], acc);
    }
}

// ---------------------------------------------------------------------------
// 5) output = measured @ dense_w + dense_b   ([64,256]x[256,2])
// ---------------------------------------------------------------------------
__global__ __launch_bounds__(128) void dense_kernel(
    const float* __restrict__ measured, const float* __restrict__ dw,
    const float* __restrict__ db, float* __restrict__ out)
{
    const int i = threadIdx.x;                  // 128 = 64 b * 2 j
    const int b = i >> 1, j = i & 1;
    float acc = db[j];
#pragma unroll 4
    for (int u = 0; u < NU; ++u)
        acc = fmaf(measured[b * NU + u], dw[u * 2 + j], acc);
    out[b * 2 + j] = acc;
}

// ---------------------------------------------------------------------------
// 6) senti_out = phase[token] @ senti_w + senti_b  ([16384,512]x[512,18])
// ---------------------------------------------------------------------------
__global__ __launch_bounds__(256) void senti_kernel(
    const int* __restrict__ seq, const float* __restrict__ phT,
    const float* __restrict__ sw, const float* __restrict__ sb,
    float* __restrict__ senti_out)
{
    const int wave = threadIdx.x >> 5;
    const int lane = threadIdx.x & 31;
    const int t = blockIdx.x * 8 + wave;
    const int tok = seq[t];
    const float* prow = phT + (size_t)tok * ND;

    float acc[18];
#pragma unroll
    for (int j = 0; j < 18; ++j) acc[j] = 0.0f;

    for (int d = lane; d < ND; d += 32) {
        const float p = prow[d];
        const float* wr = sw + d * 18;
#pragma unroll
        for (int j = 0; j < 18; ++j) acc[j] = fmaf(p, wr[j], acc[j]);
    }
#pragma unroll
    for (int j = 0; j < 18; ++j)
#pragma unroll
        for (int o = 16; o > 0; o >>= 1) acc[j] += __shfl_xor(acc[j], o, 32);

    if (lane == 0) {
        float* dst = senti_out + (size_t)t * 18;
#pragma unroll
        for (int j = 0; j < 18; ++j) dst[j] = acc[j] + sb[j];
    }
}

// ---------------------------------------------------------------------------
// Launch
// ---------------------------------------------------------------------------
extern "C" void kernel_launch(void* const* d_in, const int* in_sizes, int n_in,
                              void* d_out, int out_size, void* d_ws, size_t ws_size,
                              hipStream_t stream)
{
    (void)in_sizes; (void)n_in; (void)out_size; (void)ws_size;

    const int*   seq  = (const int*)d_in[0];
    const int*   lex  = (const int*)d_in[1];
    const float* ampT = (const float*)d_in[2];
    const float* phT  = (const float*)d_in[3];
    const float* mk   = (const float*)d_in[4];
    const float* dw   = (const float*)d_in[5];
    const float* db   = (const float*)d_in[6];
    const float* sw   = (const float*)d_in[7];
    const float* sb   = (const float*)d_in[8];

    // Workspace layout (bytes)
    char* ws = (char*)d_ws;
    __bf16* RI       = (__bf16*)(ws);                                   // 32 MB
    __bf16* Kr       = (__bf16*)(ws + (size_t)NTOK * LDRI * 2);         // 256 KB
    __bf16* Ki       = (__bf16*)(ws + (size_t)NTOK * LDRI * 2 + (size_t)NU * ND * 2);
    float*  norms    = (float*)(ws + (size_t)NTOK * LDRI * 2 + (size_t)NU * ND * 4);
    float*  wsm      = norms + NTOK;
    float*  measured = wsm + NTOK;

    // Output layout: senti_out [16384*18] f32 | senti_tag [16384] i64 | output [64*2] f32
    float*     senti_out = (float*)d_out;
    long long* tag_out   = (long long*)(senti_out + (size_t)NTOK * 18);
    float*     out       = (float*)(tag_out + NTOK);

    embed_kernel<<<NTOK, 256, 0, stream>>>(seq, lex, ampT, phT, RI, norms, tag_out);
    softmax_kernel<<<NB, 256, 0, stream>>>(norms, wsm, measured);
    prepk_kernel<<<NU, 256, 0, stream>>>(mk, Kr, Ki);
    qmeasure_wmma_kernel<<<dim3(NTOK / 16, NU / 64), 128, 0, stream>>>(RI, Kr, Ki, wsm, measured);
    dense_kernel<<<1, 128, 0, stream>>>(measured, dw, db, out);
    senti_kernel<<<NTOK / 8, 256, 0, stream>>>(seq, phT, sw, sb, senti_out);
}
